// Policy_16621523435651
// MI455X (gfx1250) — matrix operand order repaired and checked by reading, not compiled, not run on hardware
//
#include <hip/hip_runtime.h>
#include <cstdint>
#include <cstddef>

typedef __attribute__((ext_vector_type(16))) __bf16 v16bf;
typedef __attribute__((ext_vector_type(8)))  float  v8f;
typedef __attribute__((ext_vector_type(4)))  unsigned int u32x4;
typedef __attribute__((ext_vector_type(8)))  unsigned int u32x8;

namespace {

constexpr int   kFIn    = 256;
constexpr int   kFH     = 512;
constexpr int   kNA     = 40;
constexpr int   kNB     = 4;
constexpr int   kGraphs = 4096;
constexpr int   kNPG    = 32;                       // nodes per graph
constexpr int   kNodes  = kGraphs * kNPG;           // 131072
constexpr int   kDOut   = kNB + kNB * kNA;          // 164
constexpr int   kDPad   = 176;                      // 164 padded to 16
constexpr float kEps    = 1e-5f;

// LDS staging for GEMM1 B-slice: 256 rows x 512 elems, padded row stride 520
// (TDM pads 4 dwords after every 256 dwords -> bank step 4 for ds_load_b128)
constexpr int    kBRowLds   = 520;
constexpr size_t kLdsBytes  = (size_t)256 * kBRowLds * 2;   // 266,240 B (< 320KB)

// ---- output layout (flat, f32): append | connect | end ----
constexpr size_t kAppendElems = (size_t)kNodes * (kNA * kNB);   // 20,971,520
constexpr size_t kConnectOff  = kAppendElems;
constexpr size_t kEndOff      = kAppendElems + (size_t)kNodes * kNB; // 21,495,808

// ---- workspace offsets (bytes, all 256-aligned) ----
constexpr size_t OFF_XB     = 0;                                  // bf16 X        131072x256
constexpr size_t OFF_XENDF  = OFF_XB     + (size_t)kNodes*kFIn*2; // f32  X_end    4096x256
constexpr size_t OFF_XENDB  = OFF_XENDF  + (size_t)kGraphs*kFIn*4;// bf16 X_end
constexpr size_t OFF_WHB    = OFF_XENDB  + (size_t)kGraphs*kFIn*2;// bf16 W_h      512x512
constexpr size_t OFF_WXB    = OFF_WHB    + (size_t)kFH*kFH*2;     // bf16 W_x pad  176x512
constexpr size_t OFF_XHPRE  = OFF_WXB    + (size_t)kDPad*kFH*2;   // bf16 Xh_pre   131072x512
constexpr size_t OFF_XX     = OFF_XHPRE  + (size_t)kNodes*kFH*2;  // f32  X_x      131072x164
constexpr size_t OFF_XHEPRE = OFF_XX     + (size_t)kNodes*kDOut*4;// f32  Xhe_pre  4096x512
constexpr size_t OFF_P1     = OFF_XHEPRE + (size_t)kGraphs*kFH*4; // f32 partials  512 blk x 1024
constexpr size_t OFF_P2     = OFF_P1     + (size_t)512*1024*4;    // f32 partials  64 blk x 1024
constexpr size_t OFF_SS1    = OFF_P2     + (size_t)64*1024*4;     // f32 scale[512]|shift[512]
constexpr size_t OFF_SS2    = OFF_SS1    + 4096;
constexpr size_t OFF_XXE    = OFF_SS2    + 4096;                  // f32 4096
constexpr size_t OFF_XSUM   = OFF_XXE    + (size_t)kGraphs*4;     // f32 4096 (stores 1/sum)

// ------------------------------------------------------------------
// WMMA fragment loaders (bf16, per CDNA5 ISA VGPR layouts)
// A 16x32 (MxK): lane l: m=l&15, khalf=l>>4; elems i<8 -> k=khalf*8+i,
//                i>=8 -> k=16+khalf*8+(i-8): two contiguous 16B chunks.
// B 32x16 (KxN): lane l: n=l&15, half=l>>4; elems i<8 -> k=half*8+i,
//                i>=8 -> k=16+half*8+(i-8): two contiguous 16B chunks.
// ------------------------------------------------------------------
__device__ __forceinline__ v16bf load_frag_a(const __bf16* p) {
  union { v16bf v; uint4 q[2]; } u;
  u.q[0] = *reinterpret_cast<const uint4*>(p);
  u.q[1] = *reinterpret_cast<const uint4*>(p + 16);
  return u.v;
}
__device__ __forceinline__ v16bf load_frag_b(const __bf16* p) {
  union { v16bf v; uint4 q[2]; } u;
  u.q[0] = *reinterpret_cast<const uint4*>(p);
  u.q[1] = *reinterpret_cast<const uint4*>(p + 16);
  return u.v;
}
__device__ __forceinline__ v8f wmma_bf16(v16bf a, v16bf b, v8f c) {
  return __builtin_amdgcn_wmma_f32_16x16x32_bf16(false, a, false, b, (short)0, c,
                                                 false, false);
}

// ------------------------------------------------------------------ kernels
__global__ __launch_bounds__(256) void kconv_x(const float* __restrict__ X,
                                               __bf16* __restrict__ Xb) {
  size_t i = ((size_t)blockIdx.x * 256 + threadIdx.x) * 4;
  float4 v = *reinterpret_cast<const float4*>(X + i);
  Xb[i + 0] = (__bf16)v.x;  Xb[i + 1] = (__bf16)v.y;
  Xb[i + 2] = (__bf16)v.z;  Xb[i + 3] = (__bf16)v.w;
}

__global__ __launch_bounds__(256) void kxend(const float* __restrict__ X,
                                             float* __restrict__ XendF,
                                             __bf16* __restrict__ XendB) {
  int g = blockIdx.x, t = threadIdx.x;
  const float* p = X + (size_t)g * kNPG * kFIn + t;
  float s = 0.f;
#pragma unroll
  for (int n = 0; n < kNPG; ++n) s += p[(size_t)n * kFIn];
  float m = s * (1.0f / kNPG);
  XendF[(size_t)g * kFIn + t] = m;
  XendB[(size_t)g * kFIn + t] = (__bf16)m;
}

__global__ __launch_bounds__(256) void kconv_w(const float* __restrict__ W_h,
                                               const float* __restrict__ W_x,
                                               __bf16* __restrict__ WhB,
                                               __bf16* __restrict__ WxB) {
  int i = blockIdx.x * 256 + threadIdx.x;
  if (i < kFH * kFH) {
    WhB[i] = (__bf16)W_h[i];
  } else {
    int j = i - kFH * kFH;                 // 0 .. 176*512-1
    WxB[j] = (j < kDOut * kFH) ? (__bf16)W_x[j] : (__bf16)0.0f;
  }
}

// GEMM1: Xh_pre[131072,512] = [X | Xend] @ W_h^T   (bf16 WMMA, f32 acc)
// B-slice (256 rows of W_h x K=512) staged into LDS by the Tensor Data Mover;
// block: 8 waves; wave -> 64 rows x 32 cols (4 M-subtiles x 2 N-subtiles)
// grid: (M/64, N/256) = (2048, 2); dynamic LDS = kLdsBytes
__global__ __launch_bounds__(256) void kgemm1(const __bf16* __restrict__ Xb,
                                              const __bf16* __restrict__ XendB,
                                              const __bf16* __restrict__ WhB,
                                              __bf16* __restrict__ XhPre) {
  extern __shared__ char smem[];
  __bf16* Blds = reinterpret_cast<__bf16*>(smem);

  const int wid = threadIdx.x >> 5;
  const int l   = threadIdx.x & 31;
  const int mr  = l & 15;
  const int kh  = l >> 4;
  const int m0  = blockIdx.x * 64;
  const int nw0 = wid * 32;                 // wave's column base within LDS slice
  const int n0g = blockIdx.y * 256;         // block's column base in W_h

  // --- TDM: DMA W_h[n0g .. n0g+255][0..511] (bf16) into LDS, padded rows ---
  if (wid == 0) {
    unsigned lds_base = (unsigned)(size_t)(void*)Blds;
    unsigned long long ga =
        (unsigned long long)(size_t)(WhB + (size_t)n0g * kFH);
    u32x4 g0;
    g0[0] = 1u;                                   // count=1, user descriptor
    g0[1] = lds_base;                             // lds_addr
    g0[2] = (unsigned)ga;                         // global_addr[31:0]
    g0[3] = (unsigned)((ga >> 32) & 0x1FFFFFFu)   // global_addr[56:32]
          | (2u << 30);                           // type=2 ("image")
    u32x8 g1;
    g1[0] = (1u << 16)        // data_size = 1 -> 2 bytes
          | (1u << 20)        // pad_enable
          | (7u << 22)        // pad_interval: 256 dwords (= one 512-elem row)
          | (3u << 25);       // pad_amount: 4 dwords (row stride 520 elems)
    g1[1] = ((unsigned)kFH & 0xFFFFu) << 16;      // tensor_dim0[15:0]=512
    g1[2] = (256u & 0xFFFFu) << 16;               // dim0 hi=0 | tensor_dim1[15:0]=256
    g1[3] = ((unsigned)kFH) << 16;                // dim1 hi=0 | tile_dim0=512
    g1[4] = 256u;                                 // tile_dim1=256 | tile_dim2=0
    g1[5] = (unsigned)kFH;                        // tensor_dim0_stride=512
    g1[6] = 0u;                                   // stride hi | tensor_dim1_stride lo
    g1[7] = 0u;                                   // tensor_dim1_stride hi
    u32x4 gz = (u32x4){0u, 0u, 0u, 0u};           // groups 2/3 unused (2D tile)
    asm volatile("tensor_load_to_lds %0, %1, %2, %3"
                 :: "s"(g0), "s"(g1), "s"(gz), "s"(gz)
                 : "memory");
    __builtin_amdgcn_s_wait_tensorcnt(0);
  }
  __syncthreads();

  v8f acc[4][2];
#pragma unroll
  for (int s = 0; s < 4; ++s)
#pragma unroll
    for (int t = 0; t < 2; ++t) acc[s][t] = (v8f){0, 0, 0, 0, 0, 0, 0, 0};

#pragma unroll 4
  for (int kt = 0; kt < 16; ++kt) {
    const int k0 = kt * 32;
    v16bf a[4];
#pragma unroll
    for (int s = 0; s < 4; ++s) {
      const int row = m0 + s * 16 + mr;
      if (k0 < kFIn)   // k in X region (k0 multiple of 32, whole step inside)
        a[s] = load_frag_a(Xb + (size_t)row * kFIn + k0 + kh * 8);
      else             // k in broadcast X_end region
        a[s] = load_frag_a(XendB + (size_t)(row >> 5) * kFIn + (k0 - kFIn) + kh * 8);
    }
    v16bf b[2];
#pragma unroll
    for (int t = 0; t < 2; ++t)
      b[t] = load_frag_b(Blds + (size_t)(nw0 + t * 16 + mr) * kBRowLds + k0 + kh * 8);
#pragma unroll
    for (int s = 0; s < 4; ++s)
#pragma unroll
      for (int t = 0; t < 2; ++t) acc[s][t] = wmma_bf16(a[s], b[t], acc[s][t]);
  }

#pragma unroll
  for (int s = 0; s < 4; ++s)
#pragma unroll
    for (int t = 0; t < 2; ++t) {
      const int n = n0g + nw0 + t * 16 + mr;
#pragma unroll
      for (int r = 0; r < 8; ++r) {
        const int m = m0 + s * 16 + r + 8 * kh;
        XhPre[(size_t)m * kFH + n] = (__bf16)acc[s][t][r];
      }
    }
}

// per-block column sum/sumsq partials over 256 rows of Xh_pre (bf16)
__global__ __launch_bounds__(512) void kstats_bf16(const __bf16* __restrict__ A,
                                                   float* __restrict__ P) {
  const int t  = threadIdx.x;             // column 0..511
  const int r0 = blockIdx.x * 256;
  float s = 0.f, s2 = 0.f;
  for (int r = 0; r < 256; ++r) {
    float v = (float)A[(size_t)(r0 + r) * kFH + t];
    s += v; s2 += v * v;
  }
  P[(size_t)blockIdx.x * 1024 + t]       = s;
  P[(size_t)blockIdx.x * 1024 + 512 + t] = s2;
}

// per-block column partials over 64 rows of f32 matrix (end path)
__global__ __launch_bounds__(512) void kstats_f32(const float* __restrict__ A,
                                                  float* __restrict__ P) {
  const int t  = threadIdx.x;
  const int r0 = blockIdx.x * 64;
  float s = 0.f, s2 = 0.f;
  for (int r = 0; r < 64; ++r) {
    float v = A[(size_t)(r0 + r) * kFH + t];
    s += v; s2 += v * v;
  }
  P[(size_t)blockIdx.x * 1024 + t]       = s;
  P[(size_t)blockIdx.x * 1024 + 512 + t] = s2;
}

// reduce partials + finalize BN affine: ss = [gamma*rstd | beta - mean*gamma*rstd]
__global__ __launch_bounds__(512) void kred_fin(const float* __restrict__ P, int nblk,
                                                const float* __restrict__ g,
                                                const float* __restrict__ b,
                                                float inv_cnt, float* __restrict__ ss) {
  const int t = threadIdx.x;
  float s = 0.f, s2 = 0.f;
  for (int bb = 0; bb < nblk; ++bb) {
    s  += P[(size_t)bb * 1024 + t];
    s2 += P[(size_t)bb * 1024 + 512 + t];
  }
  float m    = s * inv_cnt;
  float var  = s2 * inv_cnt - m * m;
  float rstd = rsqrtf(var + kEps);
  float sc   = g[t] * rstd;
  ss[t]       = sc;
  ss[512 + t] = b[t] - m * sc;
}

// in-place BN + ReLU on Xh_pre (bf16)
__global__ __launch_bounds__(256) void kbn_apply(__bf16* __restrict__ A,
                                                 const float* __restrict__ ss) {
  size_t base = ((size_t)blockIdx.x * 256 + threadIdx.x) * 4;
#pragma unroll
  for (int j = 0; j < 4; ++j) {
    size_t i = base + j;
    int c = (int)(i & (kFH - 1));
    float x = (float)A[i];
    float y = fmaxf(0.f, x * ss[c] + ss[512 + c]);
    A[i] = (__bf16)y;
  }
}

// GEMM2: X_x[131072,164] = exp(Xh @ W_x^T + b_x)   (bf16 WMMA, f32 acc)
// block: 8 waves; wave -> 32 rows x 16 cols. grid: (M/256, 176/16) = (512, 11)
__global__ __launch_bounds__(256) void kgemm2(const __bf16* __restrict__ A,
                                              const __bf16* __restrict__ B,
                                              const float* __restrict__ bias,
                                              float* __restrict__ Xx) {
  const int wid = threadIdx.x >> 5;
  const int l   = threadIdx.x & 31;
  const int mr  = l & 15;
  const int kh  = l >> 4;
  const int m0  = blockIdx.x * 256 + wid * 32;
  const int n0  = blockIdx.y * 16;

  v8f acc[2];
  acc[0] = (v8f){0, 0, 0, 0, 0, 0, 0, 0};
  acc[1] = (v8f){0, 0, 0, 0, 0, 0, 0, 0};

#pragma unroll 4
  for (int kt = 0; kt < 16; ++kt) {
    const int k0 = kt * 32;
    v16bf a0 = load_frag_a(A + (size_t)(m0 + mr) * kFH + k0 + kh * 8);
    v16bf a1 = load_frag_a(A + (size_t)(m0 + 16 + mr) * kFH + k0 + kh * 8);
    v16bf bq = load_frag_b(B + (size_t)(n0 + mr) * kFH + k0 + kh * 8);
    acc[0] = wmma_bf16(a0, bq, acc[0]);
    acc[1] = wmma_bf16(a1, bq, acc[1]);
  }

  const int n = n0 + mr;
  if (n < kDOut) {
    const float bv = bias[n];
#pragma unroll
    for (int s = 0; s < 2; ++s)
#pragma unroll
      for (int r = 0; r < 8; ++r) {
        const int m = m0 + s * 16 + r + 8 * kh;
        Xx[(size_t)m * kDOut + n] = expf(acc[s][r] + bv);
      }
  }
}

// end path: Xhe_pre[4096,512] = X_end @ W_ht^T (small, plain VALU)
__global__ __launch_bounds__(512) void kend_gemm(const float* __restrict__ XendF,
                                                 const float* __restrict__ Wht,
                                                 float* __restrict__ XhePre) {
  __shared__ float xr[kFIn];
  const int g = blockIdx.x, t = threadIdx.x;
  if (t < kFIn) xr[t] = XendF[(size_t)g * kFIn + t];
  __syncthreads();
  const float* w = Wht + (size_t)t * kFIn;
  float acc = 0.f;
#pragma unroll 4
  for (int k = 0; k < kFIn; ++k) acc += xr[k] * w[k];
  XhePre[(size_t)g * kFH + t] = acc;
}

// X_x_end[g] = exp(dot(relu(BN(Xhe_pre[g])), W_xt) + b_xt)
__global__ __launch_bounds__(512) void kxxe(const float* __restrict__ XhePre,
                                            const float* __restrict__ ss,
                                            const float* __restrict__ Wxt,
                                            const float* __restrict__ bxt,
                                            float* __restrict__ Xxe) {
  __shared__ float sd[512];
  const int g = blockIdx.x, t = threadIdx.x;
  float h = fmaxf(0.f, XhePre[(size_t)g * kFH + t] * ss[t] + ss[512 + t]);
  sd[t] = h * Wxt[t];
  __syncthreads();
  for (int s = 256; s > 0; s >>= 1) {
    if (t < s) sd[t] += sd[t + s];
    __syncthreads();
  }
  if (t == 0) Xxe[g] = expf(sd[0] + bxt[0]);
}

// per-graph: sum(X_x over 32x164) + Xxe -> store 1/sum; write `end` output
__global__ __launch_bounds__(256) void kxsum(const float* __restrict__ Xx,
                                             const float* __restrict__ Xxe,
                                             float* __restrict__ XsumInv,
                                             float* __restrict__ out) {
  __shared__ float sd[256];
  const int g = blockIdx.x, t = threadIdx.x;
  const float* p = Xx + (size_t)g * kNPG * kDOut;
  float s = 0.f;
  for (int i = t; i < kNPG * kDOut; i += 256) s += p[i];
  sd[t] = s;
  __syncthreads();
  for (int w = 128; w > 0; w >>= 1) {
    if (t < w) sd[t] += sd[t + w];
    __syncthreads();
  }
  if (t == 0) {
    float e   = Xxe[g];
    float inv = 1.0f / (sd[0] + e);
    XsumInv[g]         = inv;
    out[kEndOff + g]   = e * inv;
  }
}

// normalize + scatter outputs: append (cols 4..163) and connect (cols 0..3)
__global__ __launch_bounds__(256) void kout(const float* __restrict__ Xx,
                                            const float* __restrict__ XsumInv,
                                            float* __restrict__ out) {
  size_t base = ((size_t)blockIdx.x * 256 + threadIdx.x) * 4;
#pragma unroll
  for (int j = 0; j < 4; ++j) {
    size_t i = base + j;
    int node = (int)(i / kDOut);
    int f    = (int)(i - (size_t)node * kDOut);
    float v  = Xx[i] * XsumInv[node >> 5];
    if (f < kNB) out[kConnectOff + (size_t)node * kNB + f] = v;
    else         out[(size_t)node * (kNA * kNB) + (f - kNB)] = v;
  }
}

}  // namespace

extern "C" void kernel_launch(void* const* d_in, const int* in_sizes, int n_in,
                              void* d_out, int out_size, void* d_ws, size_t ws_size,
                              hipStream_t stream) {
  const float* X    = (const float*)d_in[0];
  const float* W_h  = (const float*)d_in[3];
  const float* g_h  = (const float*)d_in[4];
  const float* b_h  = (const float*)d_in[5];
  const float* W_ht = (const float*)d_in[6];
  const float* g_ht = (const float*)d_in[7];
  const float* b_ht = (const float*)d_in[8];
  const float* W_x  = (const float*)d_in[9];
  const float* b_x  = (const float*)d_in[10];
  const float* W_xt = (const float*)d_in[11];
  const float* b_xt = (const float*)d_in[12];
  float* out = (float*)d_out;
  char*  ws  = (char*)d_ws;

  __bf16* Xb     = (__bf16*)(ws + OFF_XB);
  float*  XendF  = (float*)(ws + OFF_XENDF);
  __bf16* XendB  = (__bf16*)(ws + OFF_XENDB);
  __bf16* WhB    = (__bf16*)(ws + OFF_WHB);
  __bf16* WxB    = (__bf16*)(ws + OFF_WXB);
  __bf16* XhPre  = (__bf16*)(ws + OFF_XHPRE);
  float*  Xx     = (float*)(ws + OFF_XX);
  float*  XhePre = (float*)(ws + OFF_XHEPRE);
  float*  P1     = (float*)(ws + OFF_P1);
  float*  P2     = (float*)(ws + OFF_P2);
  float*  SS1    = (float*)(ws + OFF_SS1);
  float*  SS2    = (float*)(ws + OFF_SS2);
  float*  Xxe    = (float*)(ws + OFF_XXE);
  float*  Xsum   = (float*)(ws + OFF_XSUM);

  // 1) fp32 -> bf16 conversions + per-graph mean
  kconv_x<<<32768, 256, 0, stream>>>(X, Xb);
  kxend<<<kGraphs, 256, 0, stream>>>(X, XendF, XendB);
  kconv_w<<<(kFH * kFH + kDPad * kFH) / 256, 256, 0, stream>>>(W_h, W_x, WhB, WxB);

  // 2) GEMM1 (bf16 WMMA, B staged via Tensor Data Mover) -> Xh_pre
  kgemm1<<<dim3(kNodes / 64, kFH / 256), 256, kLdsBytes, stream>>>(Xb, XendB, WhB,
                                                                   XhPre);

  // 3) BatchNorm stats (two-stage, deterministic) + in-place BN+ReLU
  kstats_bf16<<<512, 512, 0, stream>>>(XhPre, P1);
  kred_fin<<<1, 512, 0, stream>>>(P1, 512, g_h, b_h, 1.0f / (float)kNodes, SS1);
  kbn_apply<<<65536, 256, 0, stream>>>(XhPre, SS1);

  // 4) GEMM2 (bf16 WMMA) with exp epilogue -> X_x
  kgemm2<<<dim3(kNodes / 256, kDPad / 16), 256, 0, stream>>>(XhPre, WxB, b_x, Xx);

  // 5) small end path
  kend_gemm<<<kGraphs, 512, 0, stream>>>(XendF, W_ht, XhePre);
  kstats_f32<<<64, 512, 0, stream>>>(XhePre, P2);
  kred_fin<<<1, 512, 0, stream>>>(P2, 64, g_ht, b_ht, 1.0f / (float)kGraphs, SS2);
  kxxe<<<kGraphs, 512, 0, stream>>>(XhePre, SS2, W_xt, b_xt, Xxe);

  // 6) per-graph softmax denominator + outputs
  kxsum<<<kGraphs, 256, 0, stream>>>(Xx, Xxe, Xsum, out);
  kout<<<20992, 256, 0, stream>>>(Xx, Xsum, out);

  (void)in_sizes; (void)n_in; (void)out_size; (void)ws_size;
}
